// InducedFieldBlockV32_4312147165457
// MI455X (gfx1250) — matrix operand order repaired and checked by reading, not compiled, not run on hardware
//
#include <hip/hip_runtime.h>
#include <hip/hip_bf16.h>
#include <cstring>

typedef __bf16 bf16_t;
typedef __attribute__((ext_vector_type(16))) __bf16 v16bf;
typedef __attribute__((ext_vector_type(8)))  __bf16 v8bf;
typedef __attribute__((ext_vector_type(8)))  float  v8f;

#define DMODEL 1024
#define NHEADS 16
#define HDIM   64
#define FFDIM  4096
#define BATCH  8
#define NPTS   1024
#define NCTX   256
#define MIND   64
#define LKV1   1280
#define LKV2   320

// ---------------- WMMA fragment loaders (bf16, 16x16x32) ----------------
// A (16xK): lane half 0 -> K {k..k+7, k+16..k+23}; half 1 -> {k+8..k+15, k+24..k+31}
__device__ __forceinline__ v16bf load_a_frag(const bf16_t* row, int k, int half) {
  union { v16bf v; v8bf h[2]; } u;
  const bf16_t* p = row + k + half * 8;
  u.h[0] = *(const v8bf*)(p);
  u.h[1] = *(const v8bf*)(p + 16);
  return u.v;
}
// B (Kx16) consumed as BT[N][K]: lane half 0 -> K {k..k+7, k+16..k+23}; half 1 -> {k+8..k+15, k+24..k+31}
__device__ __forceinline__ v16bf load_b_frag(const bf16_t* col, int k, int half) {
  union { v16bf v; v8bf h[2]; } u;
  const bf16_t* p = col + k + half * 8;
  u.h[0] = *(const v8bf*)(p);
  u.h[1] = *(const v8bf*)(p + 16);
  return u.v;
}

// ---------------- generic batched GEMM: D = act(alpha*A@B + bias) (+Res) ----------------
struct GemmP {
  const bf16_t* A; const bf16_t* B;
  const float* bias; const float* Res;
  float* Df; bf16_t* Db; bf16_t* Vt;
  long long sAb, sAh, sBb, sBh, sDfb, sDfh, sDbb, sDbh, sRb, sVtb;
  int lda, ldb, lddf, lddb, ldr, M, N, K, H, outL, flags; // flags bit0 = exact GELU
  float alpha;
};

__global__ __launch_bounds__(128) void gemm_bf16_wmma(GemmP p) {
  const int lane = threadIdx.x & 31;
  const int wave = threadIdx.x >> 5;
  const int half = lane >> 4;
  const int l16  = lane & 15;
  const int z  = blockIdx.z;
  const int bb = z / p.H;
  const int hh = z - bb * p.H;
  const int m0 = blockIdx.y * 64 + wave * 16;
  const int n0 = blockIdx.x * 64;
  if (m0 >= p.M) return;

  const bf16_t* arow  = p.A + bb * p.sAb + hh * p.sAh + (long long)(m0 + l16) * p.lda;
  const bf16_t* Bbase = p.B + bb * p.sBb + hh * p.sBh;
  const bf16_t* brow0 = Bbase + (long long)(n0 +  0 + l16) * p.ldb;
  const bf16_t* brow1 = Bbase + (long long)(n0 + 16 + l16) * p.ldb;
  const bf16_t* brow2 = Bbase + (long long)(n0 + 32 + l16) * p.ldb;
  const bf16_t* brow3 = Bbase + (long long)(n0 + 48 + l16) * p.ldb;

  v8f acc0 = {0,0,0,0,0,0,0,0};
  v8f acc1 = acc0, acc2 = acc0, acc3 = acc0;

  for (int k = 0; k < p.K; k += 32) {
    __builtin_prefetch(arow + k + 256, 0, 0);
    v16bf a  = load_a_frag(arow,  k, half);
    v16bf b0 = load_b_frag(brow0, k, half);
    v16bf b1 = load_b_frag(brow1, k, half);
    v16bf b2 = load_b_frag(brow2, k, half);
    v16bf b3 = load_b_frag(brow3, k, half);
    acc0 = __builtin_amdgcn_wmma_f32_16x16x32_bf16(false, a, false, b0, (short)0, acc0, false, false);
    acc1 = __builtin_amdgcn_wmma_f32_16x16x32_bf16(false, a, false, b1, (short)0, acc1, false, false);
    acc2 = __builtin_amdgcn_wmma_f32_16x16x32_bf16(false, a, false, b2, (short)0, acc2, false, false);
    acc3 = __builtin_amdgcn_wmma_f32_16x16x32_bf16(false, a, false, b3, (short)0, acc3, false, false);
  }

  v8f accs[4] = {acc0, acc1, acc2, acc3};
  #pragma unroll
  for (int j = 0; j < 4; ++j) {
    const int nn = n0 + j * 16 + l16;
    const float bsv = p.bias ? p.bias[nn] : 0.0f;
    #pragma unroll
    for (int r = 0; r < 8; ++r) {
      const int mm = m0 + r + 8 * half;
      float v = accs[j][r] * p.alpha + bsv;
      if (p.flags & 1) v = 0.5f * v * (1.0f + erff(v * 0.70710678118654752f));
      if (p.Res) v += p.Res[bb * p.sRb + (long long)mm * p.ldr + nn];
      if (p.Df)  p.Df[bb * p.sDfb + hh * p.sDfh + (long long)mm * p.lddf + nn] = v;
      if (p.Db)  p.Db[bb * p.sDbb + hh * p.sDbh + (long long)mm * p.lddb + nn] = (bf16_t)v;
      if (p.Vt) { // per-head transposed bf16 store: Vt[b][h][d][token]
        const int hd = nn >> 6, dd = nn & 63;
        p.Vt[bb * p.sVtb + ((long long)(hd * 64 + dd)) * p.outL + mm] = (bf16_t)v;
      }
    }
  }
}

// ---------------- elementwise / conversion kernels ----------------
__global__ void zero_f32(float* p, long long n) {
  long long i = (long long)blockIdx.x * 256 + threadIdx.x;
  if (i < n) p[i] = 0.0f;
}
__global__ void cvt_f32_to_bf16(const float* x, bf16_t* y, long long n) {
  long long i = (long long)blockIdx.x * 256 + threadIdx.x;
  if (i < n) y[i] = (bf16_t)x[i];
}
__global__ void transpose_cvt_bf16(const float* Wsrc, bf16_t* WT, int K, int N) {
  long long i = (long long)blockIdx.x * 256 + threadIdx.x;
  if (i >= (long long)K * N) return;
  int k = (int)(i / N), n = (int)(i % N);
  WT[(long long)n * K + k] = (bf16_t)Wsrc[i];
}
__global__ void build_kv1(const float* ctx, const float* pts, bf16_t* kvb) {
  long long i = (long long)blockIdx.x * 256 + threadIdx.x;
  const long long n = (long long)BATCH * LKV1 * DMODEL;
  if (i >= n) return;
  int c = (int)(i % DMODEL);
  int r = (int)((i / DMODEL) % LKV1);
  long long b = i / ((long long)LKV1 * DMODEL);
  float v = (r < NCTX) ? ctx[(b * NCTX + r) * DMODEL + c]
                       : pts[(b * NPTS + (r - NCTX)) * DMODEL + c];
  kvb[i] = (bf16_t)v;
}
__global__ void build_kv2(const float* ctx, const float* ind2, bf16_t* kv2b) {
  long long i = (long long)blockIdx.x * 256 + threadIdx.x;
  const long long n = (long long)BATCH * LKV2 * DMODEL;
  if (i >= n) return;
  int c = (int)(i % DMODEL);
  int r = (int)((i / DMODEL) % LKV2);
  long long b = i / ((long long)LKV2 * DMODEL);
  float v = (r < NCTX) ? ctx[(b * NCTX + r) * DMODEL + c]
                       : ind2[(b * MIND + (r - NCTX)) * DMODEL + c];
  kv2b[i] = (bf16_t)v;
}

// ---------------- LayerNorm: one wave per 1024-wide row ----------------
__global__ __launch_bounds__(256) void ln_rows(const float* X, const float* R, int resMod,
                                               const float* g, const float* b,
                                               float* Yf, bf16_t* Yb, int rows) {
  int row = blockIdx.x * 8 + (threadIdx.x >> 5);
  if (row >= rows) return;
  int lane = threadIdx.x & 31;
  const float* xr = X + (long long)row * DMODEL;
  const float* rr = nullptr;
  if (R) { long long rrow = resMod ? (row % resMod) : row; rr = R + rrow * DMODEL; }
  float v[32];
  float s = 0.0f;
  #pragma unroll
  for (int i = 0; i < 32; ++i) {
    float t = xr[lane + i * 32];
    if (rr) t += rr[lane + i * 32];
    v[i] = t; s += t;
  }
  for (int off = 16; off; off >>= 1) s += __shfl_xor(s, off, 32);
  float mu = s * (1.0f / DMODEL);
  float q = 0.0f;
  #pragma unroll
  for (int i = 0; i < 32; ++i) { float d = v[i] - mu; q += d * d; }
  for (int off = 16; off; off >>= 1) q += __shfl_xor(q, off, 32);
  float rs = rsqrtf(q * (1.0f / DMODEL) + 1e-5f);
  #pragma unroll
  for (int i = 0; i < 32; ++i) {
    int c = lane + i * 32;
    float y = (v[i] - mu) * rs * g[c] + b[c];
    if (Yf) Yf[(long long)row * DMODEL + c] = y;
    if (Yb) Yb[(long long)row * DMODEL + c] = (bf16_t)y;
  }
}

// ---------------- softmax: one wave per row of length PER*32 ----------------
template <int PER>
__global__ __launch_bounds__(256) void softmax_rows(const float* S, bf16_t* P, float* wavg,
                                                    int Mrows, int rowsTotal, float invH) {
  const int L = PER * 32;
  int row = blockIdx.x * 8 + (threadIdx.x >> 5);
  if (row >= rowsTotal) return;
  int lane = threadIdx.x & 31;
  const float* sr = S + (long long)row * L;
  float v[PER];
  float mx = -1e30f;
  #pragma unroll
  for (int i = 0; i < PER; ++i) { v[i] = sr[lane + i * 32]; mx = fmaxf(mx, v[i]); }
  for (int off = 16; off; off >>= 1) mx = fmaxf(mx, __shfl_xor(mx, off, 32));
  float sum = 0.0f;
  #pragma unroll
  for (int i = 0; i < PER; ++i) { v[i] = __expf(v[i] - mx); sum += v[i]; }
  for (int off = 16; off; off >>= 1) sum += __shfl_xor(sum, off, 32);
  float inv = 1.0f / sum;
  long long base = (long long)row * L;
  long long wbase = 0;
  if (wavg) {
    int bidx = row / (NHEADS * Mrows);
    int qidx = row % Mrows;
    wbase = ((long long)bidx * Mrows + qidx) * L;
  }
  #pragma unroll
  for (int i = 0; i < PER; ++i) {
    float pv = v[i] * inv;
    int c = lane + i * 32;
    P[base + c] = (bf16_t)pv;
    if (wavg) atomicAdd(&wavg[wbase + c], pv * invH);
  }
}

// ---------------- entropy of head-averaged probs ----------------
__global__ __launch_bounds__(256) void entropy_rows(const float* w2, float* out,
                                                    int L, int rowsTotal, float scale) {
  int row = blockIdx.x * 8 + (threadIdx.x >> 5);
  if (row >= rowsTotal) return;
  int lane = threadIdx.x & 31;
  float acc = 0.0f;
  for (int i = lane; i < L; i += 32) {
    float pv = w2[(long long)row * L + i];
    pv = fmaxf(pv, 1e-8f);
    acc -= pv * __logf(pv);
  }
  for (int off = 16; off; off >>= 1) acc += __shfl_xor(acc, off, 32);
  if (lane == 0) atomicAdd(out, acc * scale);
}

// ---------------- host ----------------
static inline GemmP gp0() { GemmP p; memset(&p, 0, sizeof(p)); p.alpha = 1.0f; p.H = 1; return p; }
static inline void rungemm(const GemmP& p, int batches, hipStream_t s) {
  dim3 grid((unsigned)(p.N / 64), (unsigned)((p.M + 63) / 64), (unsigned)batches);
  gemm_bf16_wmma<<<grid, dim3(128), 0, s>>>(p);
}
#define EWGRID(n) dim3((unsigned)(((n) + 255) / 256)), dim3(256), 0, stream

extern "C" void kernel_launch(void* const* d_in, const int* in_sizes, int n_in,
                              void* d_out, int out_size, void* d_ws, size_t ws_size,
                              hipStream_t stream) {
  (void)in_sizes; (void)n_in; (void)out_size; (void)ws_size;
  const float* point_tokens   = (const float*)d_in[0];
  const float* context_tokens = (const float*)d_in[1];
  const float* inducing       = (const float*)d_in[2];
  const float* ti_Wq = (const float*)d_in[3];  const float* ti_Wk = (const float*)d_in[4];
  const float* ti_Wv = (const float*)d_in[5];  const float* ti_Wo = (const float*)d_in[6];
  const float* ti_bq = (const float*)d_in[7];  const float* ti_bk = (const float*)d_in[8];
  const float* ti_bv = (const float*)d_in[9];  const float* ti_bo = (const float*)d_in[10];
  const float* tp_Wq = (const float*)d_in[11]; const float* tp_Wk = (const float*)d_in[12];
  const float* tp_Wv = (const float*)d_in[13]; const float* tp_Wo = (const float*)d_in[14];
  const float* tp_bq = (const float*)d_in[15]; const float* tp_bk = (const float*)d_in[16];
  const float* tp_bv = (const float*)d_in[17]; const float* tp_bo = (const float*)d_in[18];
  const float* iff_g = (const float*)d_in[19]; const float* iff_b = (const float*)d_in[20];
  const float* iff_W1 = (const float*)d_in[21]; const float* iff_b1 = (const float*)d_in[22];
  const float* iff_W2 = (const float*)d_in[23]; const float* iff_b2 = (const float*)d_in[24];
  const float* pff_g = (const float*)d_in[25]; const float* pff_b = (const float*)d_in[26];
  const float* pff_W1 = (const float*)d_in[27]; const float* pff_b1 = (const float*)d_in[28];
  const float* pff_W2 = (const float*)d_in[29]; const float* pff_b2 = (const float*)d_in[30];
  const float* ni_g = (const float*)d_in[31]; const float* ni_b = (const float*)d_in[32];
  const float* np_g = (const float*)d_in[33]; const float* np_b = (const float*)d_in[34];
  float* out = (float*)d_out;

  // ---- workspace carve-up ----
  char* W = (char*)d_ws;
  size_t off = 0;
  auto alloc = [&](size_t bytes) -> void* {
    void* p = W + off; off += (bytes + 255) & ~(size_t)255; return p;
  };
  bf16_t* indb   = (bf16_t*)alloc((size_t)MIND * DMODEL * 2);
  bf16_t* kvb    = (bf16_t*)alloc((size_t)BATCH * LKV1 * DMODEL * 2);
  bf16_t* kv2b   = (bf16_t*)alloc((size_t)BATCH * LKV2 * DMODEL * 2);
  bf16_t* WqTti  = (bf16_t*)alloc((size_t)DMODEL * DMODEL * 2);
  bf16_t* WkTti  = (bf16_t*)alloc((size_t)DMODEL * DMODEL * 2);
  bf16_t* WvTti  = (bf16_t*)alloc((size_t)DMODEL * DMODEL * 2);
  bf16_t* WoTti  = (bf16_t*)alloc((size_t)DMODEL * DMODEL * 2);
  bf16_t* WqTtp  = (bf16_t*)alloc((size_t)DMODEL * DMODEL * 2);
  bf16_t* WkTtp  = (bf16_t*)alloc((size_t)DMODEL * DMODEL * 2);
  bf16_t* WvTtp  = (bf16_t*)alloc((size_t)DMODEL * DMODEL * 2);
  bf16_t* WoTtp  = (bf16_t*)alloc((size_t)DMODEL * DMODEL * 2);
  bf16_t* W1Ti   = (bf16_t*)alloc((size_t)FFDIM * DMODEL * 2);
  bf16_t* W2Ti   = (bf16_t*)alloc((size_t)DMODEL * FFDIM * 2);
  bf16_t* W1Tp   = (bf16_t*)alloc((size_t)FFDIM * DMODEL * 2);
  bf16_t* W2Tp   = (bf16_t*)alloc((size_t)DMODEL * FFDIM * 2);
  bf16_t* Q1b    = (bf16_t*)alloc((size_t)MIND * DMODEL * 2);
  bf16_t* K1b    = (bf16_t*)alloc((size_t)BATCH * LKV1 * DMODEL * 2);
  bf16_t* Vt1    = (bf16_t*)alloc((size_t)BATCH * NHEADS * HDIM * LKV1 * 2);
  float*  S1     = (float*)alloc((size_t)BATCH * NHEADS * MIND * LKV1 * 4);
  bf16_t* P1b    = (bf16_t*)alloc((size_t)BATCH * NHEADS * MIND * LKV1 * 2);
  bf16_t* AO1b   = (bf16_t*)alloc((size_t)BATCH * MIND * DMODEL * 2);
  float*  ia     = (float*)alloc((size_t)BATCH * MIND * DMODEL * 4);
  float*  ind1   = (float*)alloc((size_t)BATCH * MIND * DMODEL * 4);
  bf16_t* LNib   = (bf16_t*)alloc((size_t)BATCH * MIND * DMODEL * 2);
  bf16_t* H1     = (bf16_t*)alloc((size_t)BATCH * MIND * FFDIM * 2);
  float*  ind2   = (float*)alloc((size_t)BATCH * MIND * DMODEL * 4);
  bf16_t* Q2b    = (bf16_t*)alloc((size_t)BATCH * NPTS * DMODEL * 2);
  bf16_t* K2b    = (bf16_t*)alloc((size_t)BATCH * LKV2 * DMODEL * 2);
  bf16_t* Vt2    = (bf16_t*)alloc((size_t)BATCH * NHEADS * HDIM * LKV2 * 2);
  float*  S2     = (float*)alloc((size_t)BATCH * NHEADS * NPTS * LKV2 * 4);
  bf16_t* P2b    = (bf16_t*)alloc((size_t)BATCH * NHEADS * NPTS * LKV2 * 2);
  float*  w2     = (float*)alloc((size_t)BATCH * NPTS * LKV2 * 4);
  bf16_t* AO2b   = (bf16_t*)alloc((size_t)BATCH * NPTS * DMODEL * 2);
  float*  pa     = (float*)alloc((size_t)BATCH * NPTS * DMODEL * 4);
  float*  pts1   = (float*)alloc((size_t)BATCH * NPTS * DMODEL * 4);
  bf16_t* LNpb   = (bf16_t*)alloc((size_t)BATCH * NPTS * DMODEL * 2);
  bf16_t* H2     = (bf16_t*)alloc((size_t)BATCH * NPTS * FFDIM * 2);

  // ---- stage 0: conversions ----
  cvt_f32_to_bf16<<<EWGRID((long long)MIND * DMODEL)>>>(inducing, indb, (long long)MIND * DMODEL);
  build_kv1<<<EWGRID((long long)BATCH * LKV1 * DMODEL)>>>(context_tokens, point_tokens, kvb);
  { const float* Ws[12] = {ti_Wq, ti_Wk, ti_Wv, ti_Wo, tp_Wq, tp_Wk, tp_Wv, tp_Wo, iff_W1, iff_W2, pff_W1, pff_W2};
    bf16_t* Ts[12] = {WqTti, WkTti, WvTti, WoTti, WqTtp, WkTtp, WvTtp, WoTtp, W1Ti, W2Ti, W1Tp, W2Tp};
    int Ks[12] = {DMODEL,DMODEL,DMODEL,DMODEL,DMODEL,DMODEL,DMODEL,DMODEL,DMODEL,FFDIM,DMODEL,FFDIM};
    int Ns[12] = {DMODEL,DMODEL,DMODEL,DMODEL,DMODEL,DMODEL,DMODEL,DMODEL,FFDIM,DMODEL,FFDIM,DMODEL};
    for (int i = 0; i < 12; ++i)
      transpose_cvt_bf16<<<EWGRID((long long)Ks[i] * Ns[i])>>>(Ws[i], Ts[i], Ks[i], Ns[i]);
  }

  const long long DD = (long long)DMODEL * DMODEL;

  // ---- MHA 1 (to_induced) ----
  { GemmP p = gp0(); p.A = indb; p.lda = DMODEL; p.B = WqTti; p.ldb = DMODEL;
    p.bias = ti_bq; p.M = MIND; p.N = DMODEL; p.K = DMODEL;
    p.Db = Q1b; p.lddb = DMODEL; rungemm(p, 1, stream); }                       // Q1 (shared)
  { GemmP p = gp0(); p.A = kvb; p.sAb = (long long)LKV1 * DMODEL; p.lda = DMODEL;
    p.B = WkTti; p.ldb = DMODEL; p.bias = ti_bk; p.M = LKV1; p.N = DMODEL; p.K = DMODEL;
    p.Db = K1b; p.sDbb = (long long)LKV1 * DMODEL; p.lddb = DMODEL; rungemm(p, BATCH, stream); } // K1
  { GemmP p = gp0(); p.A = kvb; p.sAb = (long long)LKV1 * DMODEL; p.lda = DMODEL;
    p.B = WvTti; p.ldb = DMODEL; p.bias = ti_bv; p.M = LKV1; p.N = DMODEL; p.K = DMODEL;
    p.Vt = Vt1; p.sVtb = (long long)NHEADS * HDIM * LKV1; p.outL = LKV1; rungemm(p, BATCH, stream); } // V1^T per head
  { GemmP p = gp0(); p.H = NHEADS; p.alpha = 0.125f;
    p.A = Q1b; p.sAb = 0; p.sAh = HDIM; p.lda = DMODEL;
    p.B = K1b; p.sBb = (long long)LKV1 * DMODEL; p.sBh = HDIM; p.ldb = DMODEL;
    p.M = MIND; p.N = LKV1; p.K = HDIM;
    p.Df = S1; p.sDfb = (long long)NHEADS * MIND * LKV1; p.sDfh = (long long)MIND * LKV1; p.lddf = LKV1;
    rungemm(p, BATCH * NHEADS, stream); }                                       // scores1
  softmax_rows<LKV1/32><<<dim3((BATCH*NHEADS*MIND + 7) / 8), dim3(256), 0, stream>>>(
      S1, P1b, nullptr, MIND, BATCH * NHEADS * MIND, 0.0f);
  { GemmP p = gp0(); p.H = NHEADS;
    p.A = P1b; p.sAb = (long long)NHEADS * MIND * LKV1; p.sAh = (long long)MIND * LKV1; p.lda = LKV1;
    p.B = Vt1; p.sBb = (long long)NHEADS * HDIM * LKV1; p.sBh = (long long)HDIM * LKV1; p.ldb = LKV1;
    p.M = MIND; p.N = HDIM; p.K = LKV1;
    p.Db = AO1b; p.sDbb = (long long)MIND * DMODEL; p.sDbh = HDIM; p.lddb = DMODEL;
    rungemm(p, BATCH * NHEADS, stream); }                                       // attn-out1
  { GemmP p = gp0(); p.A = AO1b; p.sAb = (long long)MIND * DMODEL; p.lda = DMODEL;
    p.B = WoTti; p.ldb = DMODEL; p.bias = ti_bo; p.M = MIND; p.N = DMODEL; p.K = DMODEL;
    p.Df = ia; p.sDfb = (long long)MIND * DMODEL; p.lddf = DMODEL; rungemm(p, BATCH, stream); } // Wo proj

  ln_rows<<<dim3((BATCH*MIND + 7)/8), dim3(256), 0, stream>>>(
      ia, inducing, MIND, ni_g, ni_b, ind1, nullptr, BATCH * MIND);             // induced = LN(ind + ia)
  ln_rows<<<dim3((BATCH*MIND + 7)/8), dim3(256), 0, stream>>>(
      ind1, nullptr, 0, iff_g, iff_b, nullptr, LNib, BATCH * MIND);             // iff inner LN -> bf16

  { GemmP p = gp0(); p.A = LNib; p.sAb = (long long)MIND * DMODEL; p.lda = DMODEL;
    p.B = W1Ti; p.ldb = DMODEL; p.bias = iff_b1; p.M = MIND; p.N = FFDIM; p.K = DMODEL; p.flags = 1;
    p.Db = H1; p.sDbb = (long long)MIND * FFDIM; p.lddb = FFDIM; rungemm(p, BATCH, stream); } // GELU(W1)
  { GemmP p = gp0(); p.A = H1; p.sAb = (long long)MIND * FFDIM; p.lda = FFDIM;
    p.B = W2Ti; p.ldb = FFDIM; p.bias = iff_b2; p.M = MIND; p.N = DMODEL; p.K = FFDIM;
    p.Res = ind1; p.sRb = (long long)MIND * DMODEL; p.ldr = DMODEL;
    p.Df = ind2; p.sDfb = (long long)MIND * DMODEL; p.lddf = DMODEL; rungemm(p, BATCH, stream); } // +W2

  build_kv2<<<EWGRID((long long)BATCH * LKV2 * DMODEL)>>>(context_tokens, ind2, kv2b);

  // ---- MHA 2 (to_points) ----
  { GemmP p = gp0(); p.A = kvb + (size_t)NCTX * DMODEL; p.sAb = (long long)LKV1 * DMODEL; p.lda = DMODEL;
    p.B = WqTtp; p.ldb = DMODEL; p.bias = tp_bq; p.M = NPTS; p.N = DMODEL; p.K = DMODEL;
    p.Db = Q2b; p.sDbb = (long long)NPTS * DMODEL; p.lddb = DMODEL; rungemm(p, BATCH, stream); } // Q2
  { GemmP p = gp0(); p.A = kv2b; p.sAb = (long long)LKV2 * DMODEL; p.lda = DMODEL;
    p.B = WkTtp; p.ldb = DMODEL; p.bias = tp_bk; p.M = LKV2; p.N = DMODEL; p.K = DMODEL;
    p.Db = K2b; p.sDbb = (long long)LKV2 * DMODEL; p.lddb = DMODEL; rungemm(p, BATCH, stream); } // K2
  { GemmP p = gp0(); p.A = kv2b; p.sAb = (long long)LKV2 * DMODEL; p.lda = DMODEL;
    p.B = WvTtp; p.ldb = DMODEL; p.bias = tp_bv; p.M = LKV2; p.N = DMODEL; p.K = DMODEL;
    p.Vt = Vt2; p.sVtb = (long long)NHEADS * HDIM * LKV2; p.outL = LKV2; rungemm(p, BATCH, stream); } // V2^T
  { GemmP p = gp0(); p.H = NHEADS; p.alpha = 0.125f;
    p.A = Q2b; p.sAb = (long long)NPTS * DMODEL; p.sAh = HDIM; p.lda = DMODEL;
    p.B = K2b; p.sBb = (long long)LKV2 * DMODEL; p.sBh = HDIM; p.ldb = DMODEL;
    p.M = NPTS; p.N = LKV2; p.K = HDIM;
    p.Df = S2; p.sDfb = (long long)NHEADS * NPTS * LKV2; p.sDfh = (long long)NPTS * LKV2; p.lddf = LKV2;
    rungemm(p, BATCH * NHEADS, stream); }                                       // scores2
  zero_f32<<<EWGRID((long long)BATCH * NPTS * LKV2)>>>(w2, (long long)BATCH * NPTS * LKV2);
  softmax_rows<LKV2/32><<<dim3((BATCH*NHEADS*NPTS + 7) / 8), dim3(256), 0, stream>>>(
      S2, P2b, w2, NPTS, BATCH * NHEADS * NPTS, 1.0f / NHEADS);
  { GemmP p = gp0(); p.H = NHEADS;
    p.A = P2b; p.sAb = (long long)NHEADS * NPTS * LKV2; p.sAh = (long long)NPTS * LKV2; p.lda = LKV2;
    p.B = Vt2; p.sBb = (long long)NHEADS * HDIM * LKV2; p.sBh = (long long)HDIM * LKV2; p.ldb = LKV2;
    p.M = NPTS; p.N = HDIM; p.K = LKV2;
    p.Db = AO2b; p.sDbb = (long long)NPTS * DMODEL; p.sDbh = HDIM; p.lddb = DMODEL;
    rungemm(p, BATCH * NHEADS, stream); }                                       // attn-out2
  { GemmP p = gp0(); p.A = AO2b; p.sAb = (long long)NPTS * DMODEL; p.lda = DMODEL;
    p.B = WoTtp; p.ldb = DMODEL; p.bias = tp_bo; p.M = NPTS; p.N = DMODEL; p.K = DMODEL;
    p.Df = pa; p.sDfb = (long long)NPTS * DMODEL; p.lddf = DMODEL; rungemm(p, BATCH, stream); } // Wo proj

  ln_rows<<<dim3((BATCH*NPTS + 7)/8), dim3(256), 0, stream>>>(
      pa, point_tokens, 0, np_g, np_b, pts1, nullptr, BATCH * NPTS);            // pts = LN(pts + pa)
  ln_rows<<<dim3((BATCH*NPTS + 7)/8), dim3(256), 0, stream>>>(
      pts1, nullptr, 0, pff_g, pff_b, nullptr, LNpb, BATCH * NPTS);             // pff inner LN -> bf16

  { GemmP p = gp0(); p.A = LNpb; p.sAb = (long long)NPTS * DMODEL; p.lda = DMODEL;
    p.B = W1Tp; p.ldb = DMODEL; p.bias = pff_b1; p.M = NPTS; p.N = FFDIM; p.K = DMODEL; p.flags = 1;
    p.Db = H2; p.sDbb = (long long)NPTS * FFDIM; p.lddb = FFDIM; rungemm(p, BATCH, stream); } // GELU(W1)
  { GemmP p = gp0(); p.A = H2; p.sAb = (long long)NPTS * FFDIM; p.lda = FFDIM;
    p.B = W2Tp; p.ldb = FFDIM; p.bias = pff_b2; p.M = NPTS; p.N = DMODEL; p.K = FFDIM;
    p.Res = pts1; p.sRb = (long long)NPTS * DMODEL; p.ldr = DMODEL;
    p.Df = out; p.sDfb = (long long)NPTS * DMODEL; p.lddf = DMODEL; rungemm(p, BATCH, stream); } // final pts -> d_out

  // ---- entropy side output ----
  float* ent = out + (long long)BATCH * NPTS * DMODEL;
  zero_f32<<<dim3(1), dim3(256), 0, stream>>>(ent, 1);
  entropy_rows<<<dim3((BATCH*NPTS + 7)/8), dim3(256), 0, stream>>>(
      w2, ent, LKV2, BATCH * NPTS, 1.0f / (BATCH * NPTS));
  (void)DD;
}
